// MemoryEfficientWindowAttention_41369124995836
// MI455X (gfx1250) — compile-verified, matrix operand order checked by reading, not run on hardware
//
#include <hip/hip_runtime.h>
#include <hip/hip_bf16.h>

typedef __bf16 bf16;
typedef __attribute__((ext_vector_type(16))) __bf16 bf16x16;
typedef __attribute__((ext_vector_type(8)))  float  f32x8;

#define N_TOK  64
#define DIMC   512
#define NHEAD  16
#define NWIN   64

// ---- workspace layout (bytes) ----
#define WQKV_ELEMS  (3 * DIMC * DIMC)       // 786432
#define WPROJ_ELEMS (DIMC * DIMC)           // 262144
#define BIAS_ELEMS  (NHEAD * N_TOK * N_TOK) // 65536
#define WQKV_OFF    0
#define WPROJ_OFF   (WQKV_ELEMS * 2)                  // 1572864
#define BIAS_OFF    (WPROJ_OFF + WPROJ_ELEMS * 2)     // 2097152

// ---- dynamic LDS layout (bytes) ----
#define XSS     520                    // halves per row (512 + 8 pad)
#define OSS     520
#define QS      40                     // q/k scratch row stride (halves)
#define PS      72                     // P scratch row stride (halves)
#define XS_B    0
#define OS_B    (N_TOK * XSS * 2)              // 66560
#define MK_B    (OS_B + N_TOK * OSS * 2)       // 133120
#define WV_B    (MK_B + N_TOK * N_TOK * 4)     // 149504
#define WV_STRIDE 10240                         // per-wave scratch bytes
#define SMEM_BYTES (WV_B + 8 * WV_STRIDE)       // 231424

// ------------------------------------------------------------------
__device__ __forceinline__ f32x8 wmma_bf16(bf16x16 a, bf16x16 b, f32x8 c) {
    return __builtin_amdgcn_wmma_f32_16x16x32_bf16(
        /*neg_a=*/false, a, /*neg_b=*/false, b,
        /*c_mod=*/(short)0, c, /*reuse_a=*/false, /*reuse_b=*/false);
}

__device__ __forceinline__ void zero8(f32x8& v) {
#pragma unroll
    for (int i = 0; i < 8; ++i) v[i] = 0.0f;
}

// Load one 16x32 bf16 A/B fragment given a lane-adjusted pointer:
// 8 halves at p, 8 halves at p+16 (K+16 block), matching the ISA layout.
__device__ __forceinline__ bf16x16 ldfrag(const bf16* p) {
    union { uint4 u[2]; bf16x16 v; } t;
    t.u[0] = *reinterpret_cast<const uint4*>(p);
    t.u[1] = *reinterpret_cast<const uint4*>(p + 16);
    return t.v;
}

// Fragment base pointer: lane<16 -> row row0+lane, K = k..  ; lane>=16 -> same row, K = k+8..
__device__ __forceinline__ const bf16* fp(const bf16* base, int stride, int row0,
                                          int k, int lane) {
    return base + (row0 + (lane & 15)) * stride + k + ((lane >> 4) << 3);
}

// GEMM: acc[4][NT] (64 x NT*16) += A(64x512 from LDS) * W^T,
// B-tile nt takes weight rows wrow[nt].. (bf16, row-major, K contiguous).
template <int NT>
__device__ __forceinline__ void gemm_xw(f32x8 (&acc)[4][NT], const bf16* a_base,
                                        int a_stride, const bf16* w,
                                        const int (&wrow)[NT], int lane) {
    for (int ks = 0; ks < 16; ++ks) {
        const int k = ks * 32;
        bf16x16 a[4], bw[NT];
#pragma unroll
        for (int mt = 0; mt < 4; ++mt)
            a[mt] = ldfrag(fp(a_base, a_stride, mt * 16, k, lane));
#pragma unroll
        for (int nt = 0; nt < NT; ++nt)
            bw[nt] = ldfrag(fp(w, DIMC, wrow[nt], k, lane));
#pragma unroll
        for (int mt = 0; mt < 4; ++mt)
#pragma unroll
            for (int nt = 0; nt < NT; ++nt)
                acc[mt][nt] = wmma_bf16(a[mt], bw[nt], acc[mt][nt]);
    }
}

// ------------------------------------------------------------------
// Prep: bf16-convert weights; gather relative-position bias to [H][64][64].
__global__ void prep_kernel(const float* __restrict__ wqkv,
                            const float* __restrict__ wproj,
                            const float* __restrict__ table,
                            const int* __restrict__ rpi,
                            bf16* __restrict__ wqkv_bf,
                            bf16* __restrict__ wproj_bf,
                            float* __restrict__ biasH) {
    int tid = blockIdx.x * blockDim.x + threadIdx.x;
    if (tid < WQKV_ELEMS) { wqkv_bf[tid] = (bf16)wqkv[tid]; return; }
    int t2 = tid - WQKV_ELEMS;
    if (t2 < WPROJ_ELEMS) { wproj_bf[t2] = (bf16)wproj[t2]; return; }
    int t3 = t2 - WPROJ_ELEMS;
    if (t3 < BIAS_ELEMS) {
        int h = t3 >> 12, nm = t3 & 4095;
        biasH[t3] = table[rpi[nm] * NHEAD + h];
    }
}

// ------------------------------------------------------------------
__global__ __launch_bounds__(256, 1)
void swin_attn_kernel(const float* __restrict__ x,
                      const float* __restrict__ mask,
                      const float* __restrict__ bqkv,
                      const float* __restrict__ bproj,
                      const bf16* __restrict__ wqkv,
                      const bf16* __restrict__ wproj,
                      const float* __restrict__ biasH,
                      float* __restrict__ out) {
    extern __shared__ __align__(16) char smem[];
    bf16*  xs  = (bf16*)(smem + XS_B);
    bf16*  osb = (bf16*)(smem + OS_B);
    float* mk  = (float*)(smem + MK_B);

    const int tid  = threadIdx.x;
    const int lane = tid & 31;
    const int wave = tid >> 5;
    const int b    = blockIdx.x;
    const int wdx  = b & (NWIN - 1);
    const int cl   = lane & 15;         // column within 16-wide tile
    const int hb   = (lane >> 4) << 3;  // +8 rows for hi half-wave

    bf16* qbuf = (bf16*)(smem + WV_B + wave * WV_STRIDE);
    bf16* kbuf = qbuf + N_TOK * QS;
    bf16* pbuf = qbuf;  // reused after S is formed

    // ---- stage x (fp32 -> bf16, packed b64 stores) and mask into LDS ----
    const float* xb = x + (size_t)b * N_TOK * DIMC;
    for (int i = tid * 4; i < N_TOK * DIMC; i += 256 * 4) {
        float4 f = *reinterpret_cast<const float4*>(xb + i);
        union { bf16 h[4]; uint2 u; } t;
        t.h[0] = (bf16)f.x; t.h[1] = (bf16)f.y;
        t.h[2] = (bf16)f.z; t.h[3] = (bf16)f.w;
        *reinterpret_cast<uint2*>(xs + (i >> 9) * XSS + (i & 511)) = t.u;
    }
    const float* mb = mask + (size_t)wdx * N_TOK * N_TOK;
    for (int i = tid * 4; i < N_TOK * N_TOK; i += 256 * 4)
        *reinterpret_cast<float4*>(mk + i) =
            *reinterpret_cast<const float4*>(mb + i);
    __syncthreads();

    const float scale = 0.17677669529663687f;  // 1/sqrt(32)

    for (int hi2 = 0; hi2 < 2; ++hi2) {
        const int h = wave + hi2 * 8;

        // ---- Q|K = x @ [Wq;Wk]^T, fused (shared A fragments) ----
        {
            f32x8 acc[4][4];
#pragma unroll
            for (int mt = 0; mt < 4; ++mt)
                for (int nt = 0; nt < 4; ++nt) zero8(acc[mt][nt]);
            const int wrows[4] = {h * 32, h * 32 + 16,
                                  DIMC + h * 32, DIMC + h * 32 + 16};
            gemm_xw<4>(acc, xs, XSS, wqkv, wrows, lane);
            // Q -> qbuf (bias + scale), K -> kbuf (bias)
#pragma unroll
            for (int nt = 0; nt < 2; ++nt) {
                float bq = bqkv[h * 32 + nt * 16 + cl];
                float bk = bqkv[DIMC + h * 32 + nt * 16 + cl];
#pragma unroll
                for (int mt = 0; mt < 4; ++mt)
#pragma unroll
                    for (int r = 0; r < 8; ++r) {
                        int off = (mt * 16 + r + hb) * QS + nt * 16 + cl;
                        qbuf[off] = (bf16)((acc[mt][nt][r] + bq) * scale);
                        kbuf[off] = (bf16)(acc[mt][nt + 2][r] + bk);
                    }
            }
        }

        // ---- S = Q @ K^T  (64x64, K=32) ----
        f32x8 S[4][4];
#pragma unroll
        for (int mt = 0; mt < 4; ++mt)
            for (int nt = 0; nt < 4; ++nt) zero8(S[mt][nt]);
        {
            bf16x16 qa[4], kbf[4];
#pragma unroll
            for (int mt = 0; mt < 4; ++mt) qa[mt] = ldfrag(fp(qbuf, QS, mt * 16, 0, lane));
#pragma unroll
            for (int nt = 0; nt < 4; ++nt) kbf[nt] = ldfrag(fp(kbuf, QS, nt * 16, 0, lane));
#pragma unroll
            for (int mt = 0; mt < 4; ++mt)
#pragma unroll
                for (int nt = 0; nt < 4; ++nt)
                    S[mt][nt] = wmma_bf16(qa[mt], kbf[nt], S[mt][nt]);
        }

        // ---- + relative-position bias + window mask ----
        const float* bh = biasH + h * (N_TOK * N_TOK);
#pragma unroll
        for (int mt = 0; mt < 4; ++mt)
#pragma unroll
            for (int nt = 0; nt < 4; ++nt)
#pragma unroll
                for (int r = 0; r < 8; ++r) {
                    int row = mt * 16 + r + hb, col = nt * 16 + cl;
                    S[mt][nt][r] += bh[row * 64 + col] + mk[row * 64 + col];
                }

        // ---- softmax over rows (row lives in one VGPR across 16 lanes) ----
#pragma unroll
        for (int mt = 0; mt < 4; ++mt)
#pragma unroll
            for (int r = 0; r < 8; ++r) {
                float m = S[mt][0][r];
#pragma unroll
                for (int nt = 1; nt < 4; ++nt) m = fmaxf(m, S[mt][nt][r]);
#pragma unroll
                for (int off = 1; off < 16; off <<= 1)
                    m = fmaxf(m, __shfl_xor(m, off, 32));
                float s = 0.0f;
#pragma unroll
                for (int nt = 0; nt < 4; ++nt) {
                    float e = __expf(S[mt][nt][r] - m);
                    S[mt][nt][r] = e;
                    s += e;
                }
#pragma unroll
                for (int off = 1; off < 16; off <<= 1)
                    s += __shfl_xor(s, off, 32);
                float inv = 1.0f / s;
#pragma unroll
                for (int nt = 0; nt < 4; ++nt) S[mt][nt][r] *= inv;
            }

        // ---- P -> LDS (bf16) for A-fragment reload ----
#pragma unroll
        for (int mt = 0; mt < 4; ++mt)
#pragma unroll
            for (int nt = 0; nt < 4; ++nt)
#pragma unroll
                for (int r = 0; r < 8; ++r)
                    pbuf[(mt * 16 + r + hb) * PS + nt * 16 + cl] = (bf16)S[mt][nt][r];

        // ---- V = x @ Wv^T + bv (kept in registers) ----
        f32x8 vA[4][2];
#pragma unroll
        for (int mt = 0; mt < 4; ++mt)
            for (int nt = 0; nt < 2; ++nt) zero8(vA[mt][nt]);
        {
            const int wrows[2] = {2 * DIMC + h * 32, 2 * DIMC + h * 32 + 16};
            gemm_xw<2>(vA, xs, XSS, wqkv, wrows, lane);
        }
#pragma unroll
        for (int nt = 0; nt < 2; ++nt) {
            float bv = bqkv[2 * DIMC + h * 32 + nt * 16 + cl];
#pragma unroll
            for (int mt = 0; mt < 4; ++mt)
#pragma unroll
                for (int r = 0; r < 8; ++r) vA[mt][nt][r] += bv;
        }
        // In-register repack: V C-accumulators -> B fragments (lane = dim column).
        bf16x16 vb[2][2];
#pragma unroll
        for (int ks = 0; ks < 2; ++ks)
#pragma unroll
            for (int nt = 0; nt < 2; ++nt)
#pragma unroll
                for (int j = 0; j < 8; ++j) {
                    vb[ks][nt][j]     = (bf16)vA[2 * ks][nt][j];
                    vb[ks][nt][8 + j] = (bf16)vA[2 * ks + 1][nt][j];
                }

        // ---- O = P @ V  (64x32, K=64 tokens) ----
        f32x8 O[4][2];
#pragma unroll
        for (int mt = 0; mt < 4; ++mt)
            for (int nt = 0; nt < 2; ++nt) zero8(O[mt][nt]);
#pragma unroll
        for (int ks = 0; ks < 2; ++ks) {
            bf16x16 pa[4];
#pragma unroll
            for (int mt = 0; mt < 4; ++mt)
                pa[mt] = ldfrag(fp(pbuf, PS, mt * 16, ks * 32, lane));
#pragma unroll
            for (int mt = 0; mt < 4; ++mt)
#pragma unroll
                for (int nt = 0; nt < 2; ++nt)
                    O[mt][nt] = wmma_bf16(pa[mt], vb[ks][nt], O[mt][nt]);
        }
        // ---- O -> LDS staging (bf16), columns h*32.. ----
#pragma unroll
        for (int mt = 0; mt < 4; ++mt)
#pragma unroll
            for (int nt = 0; nt < 2; ++nt)
#pragma unroll
                for (int r = 0; r < 8; ++r)
                    osb[(mt * 16 + r + hb) * OSS + h * 32 + nt * 16 + cl] =
                        (bf16)O[mt][nt][r];
    }
    __syncthreads();

    // ---- final projection: each wave computes a 64x64 output slice ----
    f32x8 C[4][4];
#pragma unroll
    for (int mt = 0; mt < 4; ++mt)
        for (int nt = 0; nt < 4; ++nt) zero8(C[mt][nt]);
    const int jb = wave * 64;
    {
        const int wrows[4] = {jb, jb + 16, jb + 32, jb + 48};
        gemm_xw<4>(C, osb, OSS, wproj, wrows, lane);
    }

    float* ob = out + (size_t)b * N_TOK * DIMC;
#pragma unroll
    for (int nt = 0; nt < 4; ++nt) {
        int col = jb + nt * 16 + cl;
        float bp = bproj[col];
#pragma unroll
        for (int mt = 0; mt < 4; ++mt)
#pragma unroll
            for (int r = 0; r < 8; ++r)
                ob[(mt * 16 + r + hb) * DIMC + col] = C[mt][nt][r] + bp;
    }
}

// ------------------------------------------------------------------
extern "C" void kernel_launch(void* const* d_in, const int* in_sizes, int n_in,
                              void* d_out, int out_size, void* d_ws, size_t ws_size,
                              hipStream_t stream) {
    const float* x     = (const float*)d_in[0];
    const float* mask  = (const float*)d_in[1];
    const float* table = (const float*)d_in[2];
    const float* Wqkv  = (const float*)d_in[3];
    const float* bqkv  = (const float*)d_in[4];
    const float* Wproj = (const float*)d_in[5];
    const float* bproj = (const float*)d_in[6];
    const int*   rpi   = (const int*)d_in[7];

    char* ws = (char*)d_ws;
    bf16*  wqkv_bf  = (bf16*)(ws + WQKV_OFF);
    bf16*  wproj_bf = (bf16*)(ws + WPROJ_OFF);
    float* biasH    = (float*)(ws + BIAS_OFF);

    const int prep_total = WQKV_ELEMS + WPROJ_ELEMS + BIAS_ELEMS;
    prep_kernel<<<(prep_total + 255) / 256, 256, 0, stream>>>(
        Wqkv, Wproj, table, rpi, wqkv_bf, wproj_bf, biasH);

    (void)hipFuncSetAttribute((const void*)swin_attn_kernel,
                              hipFuncAttributeMaxDynamicSharedMemorySize,
                              SMEM_BYTES);
    swin_attn_kernel<<<2048, 256, SMEM_BYTES, stream>>>(
        x, mask, bqkv, bproj, wqkv_bf, wproj_bf, biasH, (float*)d_out);
}
